// SSIMLoss_10797547782334
// MI455X (gfx1250) — compile-verified, matrix operand order checked by reading, not run on hardware
//
#include <hip/hip_runtime.h>
#include <hip/hip_bf16.h>

typedef __attribute__((ext_vector_type(16))) _Float16 v16h;
typedef __attribute__((ext_vector_type(8)))  _Float16 v8h;
typedef __attribute__((ext_vector_type(8)))  float    v8f;
typedef __attribute__((ext_vector_type(4)))  unsigned int v4u;

#define WAVES_PER_BLOCK 4
#define IMG_W 512
#define IMG_H 512
#define N_IMG 48            // 16 batch * 3 channels
#define TILES_PER_IMG 1024  // (512/16)^2
#define TOTAL_TILES (N_IMG * TILES_PER_IMG)   // 49152
#define N_PIX (16.0f * 3.0f * 512.0f * 512.0f)

// Normalized 11-tap Gaussian, sigma = 1.5 (precomputed; saves per-thread expf)
__device__ __constant__ float G11[11] = {
    0.00102838f, 0.00759879f, 0.03600077f, 0.10936070f, 0.21300550f,
    0.26601170f, 0.21300550f, 0.10936070f, 0.03600077f, 0.00759879f,
    0.00102838f
};

// ---------------------------------------------------------------------------
// One wave32 computes one 16x16 SSIM-map tile.
// 5 signals (p, t, p^2, t^2, p*t) are Gaussian-blurred via separable conv:
//   horizontal: Mid(16x16) = In(16x32) x Wh(32x16),  Wh[k,n] = w[k-n]
//   vertical:   Out(16x16) = Wv(16x32) x Mid(32x16), Wv[m,k] = w[k-m]
// Each pass is one V_WMMA_F32_16X16X32_F16 (K padded 26 -> 32).
// The D->B handoff between passes is a per-lane identity, done in registers.
// ---------------------------------------------------------------------------
__global__ __launch_bounds__(WAVES_PER_BLOCK * 32)
void SSIM_tile_wmma_kernel(const float* __restrict__ pred,
                           const float* __restrict__ targ,
                           float* __restrict__ partial)
{
    __shared__ __align__(16) _Float16 sig[WAVES_PER_BLOCK][5][32][32]; // halo tiles
    __shared__ float wsum[WAVES_PER_BLOCK];

    const int lane = threadIdx.x & 31;
    const int wave = threadIdx.x >> 5;
    const int tileId = blockIdx.x * WAVES_PER_BLOCK + wave;

    const int img = tileId >> 10;          // tile / 1024
    const int tin = tileId & 1023;
    const int ty = tin >> 5;
    const int tx = tin & 31;
    const long imgOff = (long)img * (IMG_W * IMG_H);
    const int cy = ty * 16;
    const int cx = tx * 16;

    // --- constant weight fragment ------------------------------------------
    // lane -> row m (A-frag) or col n (B-frag); half i -> K = (i>>3)*16 + kb + (i&7)
    const int nl = lane & 15;
    const int kb = (lane < 16) ? 0 : 8;
    v16h wf;    // serves as both Wh B-frag and Wv A-frag (identical values)
    #pragma unroll
    for (int i = 0; i < 16; ++i) {
        int k  = ((i >> 3) << 4) + kb + (i & 7);
        int dh = k - nl;
        wf[i] = (dh >= 0 && dh <= 10) ? (_Float16)G11[dh] : (_Float16)0.f;
    }

    // --- zero-fill this wave's signal tiles (conv zero padding + K pad) ----
    {
        v4u* base = (v4u*)&sig[wave][0][0][0];   // 5*32*32*2B / 16B = 640 chunks
        v4u z = {0u, 0u, 0u, 0u};
        #pragma unroll
        for (int i = 0; i < 20; ++i) base[i * 32 + lane] = z;
    }

    // --- load 26x26 halo, build 5 signals in f16 ---------------------------
    const int gx  = cx - 5 + lane;
    const bool xin = (lane < 26) && (gx >= 0) && (gx < IMG_W);
    for (int r = 0; r < 26; ++r) {
        int gy = cy - 5 + r;
        if (xin && (r + 8) < 26) {
            int gy2 = gy + 8;
            if (gy2 >= 0 && gy2 < IMG_H)
                __builtin_prefetch(&pred[imgOff + (long)gy2 * IMG_W + gx], 0, 1);
        }
        if (xin && gy >= 0 && gy < IMG_H) {
            float p = pred[imgOff + (long)gy * IMG_W + gx];
            float t = targ[imgOff + (long)gy * IMG_W + gx];
            sig[wave][0][r][lane] = (_Float16)p;
            sig[wave][1][r][lane] = (_Float16)t;
            sig[wave][2][r][lane] = (_Float16)(p * p);
            sig[wave][3][r][lane] = (_Float16)(t * t);
            sig[wave][4][r][lane] = (_Float16)(p * t);
        }
    }

    // --- 5 signals x (2 horizontal WMMA + 1 vertical WMMA), fully unrolled --
    const int ml = lane & 15;
    v8f dacc[5];
    #pragma unroll
    for (int s = 0; s < 5; ++s) {
        v8f c01[2];
        #pragma unroll
        for (int rb = 0; rb < 2; ++rb) {
            int row = rb * 16 + ml;
            v8h lo = *(const v8h*)&sig[wave][s][row][kb];
            v8h hi = *(const v8h*)&sig[wave][s][row][16 + kb];
            v16h a = __builtin_shufflevector(lo, hi,
                        0,1,2,3,4,5,6,7,8,9,10,11,12,13,14,15);
            v8f c = {};
            c01[rb] = __builtin_amdgcn_wmma_f32_16x16x32_f16(
                          false, a, false, wf, (short)0, c, false, false);
        }
        // D -> B fragment handoff is a per-lane identity: no LDS needed.
        v16h b;
        #pragma unroll
        for (int i = 0; i < 8; ++i) {
            b[i]     = (_Float16)c01[0][i];
            b[i + 8] = (_Float16)c01[1][i];
        }
        v8f c = {};
        dacc[s] = __builtin_amdgcn_wmma_f32_16x16x32_f16(
                      false, wf, false, b, (short)0, c, false, false);
    }

    // --- SSIM map + reduction ----------------------------------------------
    const float C1 = 0.0001f;   // (0.01*1.0)^2
    const float C2 = 0.0009f;   // (0.03*1.0)^2
    float lsum = 0.f;
    #pragma unroll
    for (int i = 0; i < 8; ++i) {
        float mu1 = dacc[0][i], mu2 = dacc[1][i];
        float ep2 = dacc[2][i], et2 = dacc[3][i], ept = dacc[4][i];
        float mu1s = mu1 * mu1, mu2s = mu2 * mu2, mu12 = mu1 * mu2;
        float s1  = ep2 - mu1s;
        float s2  = et2 - mu2s;
        float s12 = ept - mu12;
        float num = (2.f * mu12 + C1) * (2.f * s12 + C2);
        float den = (mu1s + mu2s + C1) * (s1 + s2 + C2);
        lsum += num / den;
    }
    #pragma unroll
    for (int off = 16; off > 0; off >>= 1)
        lsum += __shfl_xor(lsum, off, 32);
    if (lane == 0) wsum[wave] = lsum;
    __syncthreads();
    if (threadIdx.x == 0)
        partial[blockIdx.x] = wsum[0] + wsum[1] + wsum[2] + wsum[3];
}

// ---------------------------------------------------------------------------
// Deterministic final reduce: fixed-order strided sum + fixed-order tree.
// ---------------------------------------------------------------------------
__global__ __launch_bounds__(256)
void SSIM_reduce_kernel(const float* __restrict__ partial, int n,
                        float* __restrict__ out)
{
    __shared__ float sm[256];
    float s = 0.f;
    for (int i = threadIdx.x; i < n; i += 256) s += partial[i];
    sm[threadIdx.x] = s;
    __syncthreads();
    #pragma unroll
    for (int stride = 128; stride > 0; stride >>= 1) {
        if ((int)threadIdx.x < stride) sm[threadIdx.x] += sm[threadIdx.x + stride];
        __syncthreads();
    }
    if (threadIdx.x == 0) out[0] = 1.f - sm[0] / N_PIX;
}

extern "C" void kernel_launch(void* const* d_in, const int* in_sizes, int n_in,
                              void* d_out, int out_size, void* d_ws, size_t ws_size,
                              hipStream_t stream) {
    (void)in_sizes; (void)n_in; (void)out_size; (void)ws_size;
    const float* pred = (const float*)d_in[0];
    const float* targ = (const float*)d_in[1];
    float* partial = (float*)d_ws;                    // 12288 floats scratch

    const int blocks = TOTAL_TILES / WAVES_PER_BLOCK; // 12288, exact cover
    SSIM_tile_wmma_kernel<<<blocks, WAVES_PER_BLOCK * 32, 0, stream>>>(pred, targ, partial);
    SSIM_reduce_kernel<<<1, 256, 0, stream>>>(partial, blocks, (float*)d_out);
}
